// MultiHeadAttention_12223476924693
// MI455X (gfx1250) — matrix operand order repaired and checked by reading, not compiled, not run on hardware
//
#include <hip/hip_runtime.h>

// ---------------- problem constants ----------------
#define B_    2
#define S_    2048
#define D_    768
#define H_    12
#define DEP_  64
#define BH_   (B_ * H_)

// ---------------- types ----------------
typedef __attribute__((ext_vector_type(16))) __bf16       v16bf;
typedef __attribute__((ext_vector_type(8)))  float        v8f;
typedef __attribute__((ext_vector_type(4)))  unsigned int u32x4;
typedef __attribute__((ext_vector_type(4)))  float        f32x4;

union Frag {
    v16bf  v;
    __bf16 h[16];
    u32x4  q[2];
};

// hardware f32 -> bf16 (RNE) conversion; clang packs pairs into v_cvt_pk_bf16_f32
__device__ __forceinline__ void cvt8(Frag& f, int base, f32x4 a, f32x4 b) {
    f.h[base + 0] = (__bf16)a.x; f.h[base + 1] = (__bf16)a.y;
    f.h[base + 2] = (__bf16)a.z; f.h[base + 3] = (__bf16)a.w;
    f.h[base + 4] = (__bf16)b.x; f.h[base + 5] = (__bf16)b.y;
    f.h[base + 6] = (__bf16)b.z; f.h[base + 7] = (__bf16)b.w;
}

__device__ __forceinline__ v8f wmma_bf16(const Frag& a, const Frag& b, v8f c) {
    return __builtin_amdgcn_wmma_f32_16x16x32_bf16(
        /*neg_a=*/false, a.v, /*neg_b=*/false, b.v,
        /*c_mod=*/(short)0, c, /*reuse_a=*/false, /*reuse_b=*/false);
}

#define ZERO8 {0.f, 0.f, 0.f, 0.f, 0.f, 0.f, 0.f, 0.f}

// =====================================================================
// Kernel 1: QKV projections.  Y = X @ W + b, head-split to bf16 [B,H,S,64].
// M-blocked: one wave computes a 64(M) x 16(N) strip -> 4 accumulators
// sharing one (strided) B fragment per K-step.
// grid = (12, 64, 3), block = 128 (4 waves -> adjacent N tiles)
// =====================================================================
__global__ void __launch_bounds__(128)
qkv_proj_kernel(const float* __restrict__ q, const float* __restrict__ k,
                const float* __restrict__ v,
                const float* __restrict__ wq, const float* __restrict__ bq,
                const float* __restrict__ wk, const float* __restrict__ bk,
                const float* __restrict__ wv, const float* __restrict__ bv,
                __bf16* __restrict__ qkv_bf) {
    const int lane  = threadIdx.x & 31;
    const int wave  = threadIdx.x >> 5;
    const int tn    = (blockIdx.x * 4 + wave) * 16;   // 0..767
    const int tmB   = blockIdx.y * 64;                // 0..4032
    const int which = blockIdx.z;

    const float* X    = (which == 0) ? q  : (which == 1) ? k  : v;
    const float* W    = (which == 0) ? wq : (which == 1) ? wk : wv;
    const float* bias = (which == 0) ? bq : (which == 1) ? bk : bv;
    __bf16* out = qkv_bf + (size_t)which * ((size_t)B_ * H_ * S_ * DEP_);

    const int li = lane & 15, g = lane >> 4;
    const int n  = tn + li;
    const float* arow0 = X + (size_t)(tmB + li) * D_;

    v8f acc[4];
    #pragma unroll
    for (int s = 0; s < 4; ++s) acc[s] = (v8f)ZERO8;

    for (int k0 = 0; k0 < D_; k0 += 32) {
        // B fragment: lane group g holds K = k0+g*16 .. +15, column n
        Frag bF;
        const int kbb = k0 + g * 16;
        #pragma unroll
        for (int i = 0; i < 16; ++i)
            bF.h[i] = (__bf16)W[(size_t)(kbb + i) * D_ + n];
        // 4 A fragments (rows tmB+s*16 .. +15), each 4x16B loads + packed cvt
        #pragma unroll
        for (int s = 0; s < 4; ++s) {
            Frag a;
            const float* ar = arow0 + (size_t)s * 16 * D_;
            #pragma unroll
            for (int c = 0; c < 2; ++c) {
                const int kb = k0 + g * 16 + c * 8;
                cvt8(a, c * 8, *(const f32x4*)(ar + kb), *(const f32x4*)(ar + kb + 4));
            }
            acc[s] = wmma_bf16(a, bF, acc[s]);
        }
    }

    const float bval = bias[n];
    const int h = n / DEP_, dd = n % DEP_;
    #pragma unroll
    for (int s = 0; s < 4; ++s)
        #pragma unroll
        for (int r = 0; r < 8; ++r) {
            const int mm = tmB + s * 16 + r + g * 8;     // global row
            const int bb = mm >> 11, ss = mm & (S_ - 1);
            out[(((size_t)(bb * H_ + h) * S_) + ss) * DEP_ + dd] =
                (__bf16)(acc[s][r] + bval);
        }
}

// =====================================================================
// Kernel 2: raw scores = (Qh @ Kh^T) * 0.125 + mask * -1e9  -> fp32 attn
// N-blocked: one wave computes 16(M) x 64(N); A fragments loaded once,
// all fragment loads are contiguous 16B bf16 vectors.
// grid = (8, 128, 24), block = 128
// =====================================================================
__global__ void __launch_bounds__(128)
scores_kernel(const __bf16* __restrict__ qbf,
              const __bf16* __restrict__ kbf,
              const int* __restrict__ mask,
              float* __restrict__ attn) {
    const int lane = threadIdx.x & 31;
    const int wave = threadIdx.x >> 5;
    const int tnB  = (blockIdx.x * 4 + wave) * 64;   // key strip base
    const int tm   = blockIdx.y * 16;                // query tile
    const int bh   = blockIdx.z;
    const int b    = bh / H_;

    const int li = lane & 15, g = lane >> 4;
    const __bf16* qrow = qbf + ((size_t)bh * S_ + (tm + li)) * DEP_;

    Frag aF[2];
    #pragma unroll
    for (int t = 0; t < 2; ++t) {
        const int k0 = t * 32;
        aF[t].q[0] = *(const u32x4*)(qrow + k0 + g * 8);
        aF[t].q[1] = *(const u32x4*)(qrow + k0 + 16 + g * 8);
    }

    v8f acc[4];
    #pragma unroll
    for (int j = 0; j < 4; ++j) acc[j] = (v8f)ZERO8;

    #pragma unroll
    for (int j = 0; j < 4; ++j) {
        const __bf16* krow = kbf + ((size_t)bh * S_ + (tnB + j * 16 + li)) * DEP_;
        #pragma unroll
        for (int t = 0; t < 2; ++t) {
            const int k0 = t * 32;
            Frag bF;
            bF.q[0] = *(const u32x4*)(krow + k0 + g * 8);
            bF.q[1] = *(const u32x4*)(krow + k0 + 16 + g * 8);
            acc[j] = wmma_bf16(aF[t], bF, acc[j]);
        }
    }

    float* orow = attn + ((size_t)bh * S_ + tm) * S_;
    #pragma unroll
    for (int j = 0; j < 4; ++j) {
        const int n = tnB + j * 16 + li;
        const float madd = (float)mask[b * S_ + n] * -1e9f;
        #pragma unroll
        for (int r = 0; r < 8; ++r)
            orow[(size_t)(r + g * 8) * S_ + n] = acc[j][r] * 0.125f + madd;
    }
}

// =====================================================================
// Kernel 3: in-place row softmax over attn (2048 per row).
// One 256-thread block per row; LDS tree reductions (max, then sum).
// grid = 49152, block = 256
// =====================================================================
__global__ void __launch_bounds__(256)
softmax_kernel(float* __restrict__ attn) {
    float* p = attn + (size_t)blockIdx.x * S_;
    __shared__ float red[256];
    const int tid = threadIdx.x;

    float vals[8];
    float lmax = -3.0e38f;
    #pragma unroll
    for (int j = 0; j < 8; ++j) {
        vals[j] = p[tid + j * 256];
        lmax = fmaxf(lmax, vals[j]);
    }
    red[tid] = lmax; __syncthreads();
    for (int s = 128; s > 0; s >>= 1) {
        if (tid < s) red[tid] = fmaxf(red[tid], red[tid + s]);
        __syncthreads();
    }
    const float rmax = red[0];
    __syncthreads();

    float lsum = 0.f;
    #pragma unroll
    for (int j = 0; j < 8; ++j) {
        vals[j] = __expf(vals[j] - rmax);
        lsum += vals[j];
    }
    red[tid] = lsum; __syncthreads();
    for (int s = 128; s > 0; s >>= 1) {
        if (tid < s) red[tid] += red[tid + s];
        __syncthreads();
    }
    const float rinv = 1.0f / red[0];
    #pragma unroll
    for (int j = 0; j < 8; ++j)
        p[tid + j * 256] = vals[j] * rinv;
}

// =====================================================================
// Kernel 4: ctx = attn @ Vh, heads merged -> bf16 [B,S,768]
// M-blocked: one wave computes 64(M) x 16(N); 4 accumulators share the
// (strided) V-column B fragment; A rows are contiguous fp32 + packed cvt.
// The 4 waves of a block cover the full head depth (4 x 16 = 64).
// grid = (32, 24), block = 128
// =====================================================================
__global__ void __launch_bounds__(128)
ctx_kernel(const float* __restrict__ attn,
           const __bf16* __restrict__ vbf,
           __bf16* __restrict__ cbf) {
    const int lane = threadIdx.x & 31;
    const int wave = threadIdx.x >> 5;
    const int tn   = wave * 16;            // 0..63 within the head depth
    const int tmB  = blockIdx.x * 64;
    const int bh   = blockIdx.y;
    const int b = bh / H_, h = bh % H_;

    const int li = lane & 15, g = lane >> 4;
    const float* arow0 = attn + ((size_t)bh * S_ + (tmB + li)) * S_;
    const __bf16* vbase = vbf + (size_t)bh * S_ * DEP_;
    const int n = tn + li;

    v8f acc[4];
    #pragma unroll
    for (int s = 0; s < 4; ++s) acc[s] = (v8f)ZERO8;

    for (int k0 = 0; k0 < S_; k0 += 32) {
        Frag bF;
        const int kbb = k0 + g * 16;
        #pragma unroll
        for (int i = 0; i < 16; ++i)
            bF.h[i] = vbase[(size_t)(kbb + i) * DEP_ + n];
        #pragma unroll
        for (int s = 0; s < 4; ++s) {
            Frag a;
            const float* ar = arow0 + (size_t)s * 16 * S_;
            #pragma unroll
            for (int c = 0; c < 2; ++c) {
                const int kb = k0 + g * 16 + c * 8;
                cvt8(a, c * 8, *(const f32x4*)(ar + kb), *(const f32x4*)(ar + kb + 4));
            }
            acc[s] = wmma_bf16(a, bF, acc[s]);
        }
    }

    #pragma unroll
    for (int s = 0; s < 4; ++s)
        #pragma unroll
        for (int r = 0; r < 8; ++r) {
            const int mm = tmB + s * 16 + r + g * 8;
            cbf[((size_t)(b * S_) + mm) * D_ + h * DEP_ + n] = (__bf16)acc[s][r];
        }
}

// =====================================================================
// Kernel 5: out = ctx @ wo + bo  -> fp32 d_out
// M-blocked 64x16 per wave; A is bf16 (16B vector loads), B strided fp32.
// grid = (12, 64), block = 128
// =====================================================================
__global__ void __launch_bounds__(128)
out_proj_kernel(const __bf16* __restrict__ cbf,
                const float* __restrict__ wo, const float* __restrict__ bo,
                float* __restrict__ out) {
    const int lane = threadIdx.x & 31;
    const int wave = threadIdx.x >> 5;
    const int tn   = (blockIdx.x * 4 + wave) * 16;
    const int tmB  = blockIdx.y * 64;

    const int li = lane & 15, g = lane >> 4;
    const __bf16* arow0 = cbf + (size_t)(tmB + li) * D_;
    const int n = tn + li;

    v8f acc[4];
    #pragma unroll
    for (int s = 0; s < 4; ++s) acc[s] = (v8f)ZERO8;

    for (int k0 = 0; k0 < D_; k0 += 32) {
        Frag bF;
        const int kbb = k0 + g * 16;
        #pragma unroll
        for (int i = 0; i < 16; ++i)
            bF.h[i] = (__bf16)wo[(size_t)(kbb + i) * D_ + n];
        #pragma unroll
        for (int s = 0; s < 4; ++s) {
            Frag a;
            const __bf16* ar = arow0 + (size_t)s * 16 * D_;
            a.q[0] = *(const u32x4*)(ar + k0 + g * 16);
            a.q[1] = *(const u32x4*)(ar + k0 + g * 16 + 8);
            acc[s] = wmma_bf16(a, bF, acc[s]);
        }
    }

    const float bval = bo[n];
    #pragma unroll
    for (int s = 0; s < 4; ++s)
        #pragma unroll
        for (int r = 0; r < 8; ++r) {
            const int mm = tmB + s * 16 + r + g * 8;
            out[(size_t)mm * D_ + n] = acc[s][r] + bval;
        }
}

// =====================================================================
extern "C" void kernel_launch(void* const* d_in, const int* in_sizes, int n_in,
                              void* d_out, int out_size, void* d_ws, size_t ws_size,
                              hipStream_t stream) {
    (void)in_sizes; (void)n_in; (void)out_size; (void)ws_size;

    const float* q    = (const float*)d_in[0];
    const float* k    = (const float*)d_in[1];
    const float* v    = (const float*)d_in[2];
    const int*   mask = (const int*)  d_in[3];
    const float* wq   = (const float*)d_in[4];
    const float* bq   = (const float*)d_in[5];
    const float* wk   = (const float*)d_in[6];
    const float* bk   = (const float*)d_in[7];
    const float* wv   = (const float*)d_in[8];
    const float* bv   = (const float*)d_in[9];
    const float* wo   = (const float*)d_in[10];
    const float* bo   = (const float*)d_in[11];

    float* out  = (float*)d_out;                         // [B,S,D]
    float* attn = out + (size_t)B_ * S_ * D_;            // [B,H,S,S]

    // workspace: 4 bf16 tensors of B*H*S*64 elements (~25 MB total)
    const size_t one = (size_t)B_ * H_ * S_ * DEP_;
    __bf16* qbf = (__bf16*)d_ws;
    __bf16* kbf = qbf + one;
    __bf16* vbf = qbf + 2 * one;
    __bf16* cbf = qbf + 3 * one;   // ctx bf16 [B,S,D] (same element count)

    qkv_proj_kernel<<<dim3(D_ / 64, (B_ * S_) / 64, 3), 128, 0, stream>>>(
        q, k, v, wq, bq, wk, bk, wv, bv, qbf);
    scores_kernel<<<dim3(S_ / 256, S_ / 16, BH_), 128, 0, stream>>>(
        qbf, kbf, mask, attn);
    softmax_kernel<<<dim3(BH_ * S_), 256, 0, stream>>>(attn);
    ctx_kernel<<<dim3(S_ / 64, BH_), 128, 0, stream>>>(attn, vbf, cbf);
    out_proj_kernel<<<dim3(D_ / 64, (B_ * S_) / 64), 128, 0, stream>>>(
        cbf, wo, bo, out);
}
